// PeriodicGridRegularization_71640054497944
// MI455X (gfx1250) — matrix operand order repaired and checked by reading, not compiled, not run on hardware
//
#include <hip/hip_runtime.h>
#include <hip/hip_bf16.h>

// Problem constants (match reference): N = 96^2 grid points, D = 1*16*16 features.
#define NPTS  9216
#define DFEAT 256
// LDS row stride (in halfs) for the 32(K) x 128(N) B slab.
// 136 halfs = 272 B keeps every lane's 16B chunk aligned and spreads banks.
#define BPAD  136

typedef _Float16 f16;
typedef _Float16 v16h __attribute__((ext_vector_type(16)));
typedef _Float16 v8h  __attribute__((ext_vector_type(8)));
typedef float    v8f  __attribute__((ext_vector_type(8)));

// Build a v16h WMMA fragment from two aligned 16-byte chunks.
static __device__ inline v16h frag_from2(const f16* p0, const f16* p1) {
  v8h a = *(const v8h*)p0;
  v8h b = *(const v8h*)p1;
  v16h r;
#pragma unroll
  for (int i = 0; i < 8; ++i) { r[i] = a[i]; r[i + 8] = b[i]; }
  return r;
}

// ---------------------------------------------------------------------------
// Prep: split each f32 feature into f16 hi + f16 lo residual; exact f32 row
// norms sq[i] = sum_k x[i][k]^2. One block per row, one thread per feature.
// ---------------------------------------------------------------------------
__global__ __launch_bounds__(256) void prep_kernel(const float* __restrict__ img,
                                                   f16* __restrict__ xhi,
                                                   f16* __restrict__ xlo,
                                                   float* __restrict__ sq) {
  const int i = blockIdx.x;
  const int t = threadIdx.x;
  float v = img[(size_t)i * DFEAT + t];
  f16 h = (f16)v;
  float lo = v - (float)h;
  xhi[(size_t)i * DFEAT + t] = h;
  xlo[(size_t)i * DFEAT + t] = (f16)lo;
  __shared__ float s[256];
  s[t] = v * v;
  __syncthreads();
#pragma unroll
  for (int o = 128; o > 0; o >>= 1) {
    if (t < o) s[t] += s[t + o];
    __syncthreads();
  }
  if (t == 0) sq[i] = s[0];
}

// ---------------------------------------------------------------------------
// 64x64 LDS tile transpose: src[N][D] -> dst[D][N]. Coalesced on both sides.
// ---------------------------------------------------------------------------
__global__ __launch_bounds__(256) void transpose_kernel(const f16* __restrict__ src,
                                                        f16* __restrict__ dst) {
  __shared__ f16 tile[64][72];  // 72-half row stride: 16B aligned, bank-spread
  const int i0 = blockIdx.x * 64;
  const int k0 = blockIdx.y * 64;
  const int r  = threadIdx.x >> 2;
  const int cg = (threadIdx.x & 3) * 16;
  const v8h* g = (const v8h*)(src + (size_t)(i0 + r) * DFEAT + k0 + cg);
  v8h a = g[0], b = g[1];
  *(v8h*)&tile[r][cg]     = a;
  *(v8h*)&tile[r][cg + 8] = b;
  __syncthreads();
  const int c  = threadIdx.x >> 2;
  const int rg = (threadIdx.x & 3) * 16;
  v8h t0, t1;
#pragma unroll
  for (int q = 0; q < 8; ++q) {
    t0[q] = tile[rg + q][c];
    t1[q] = tile[rg + 8 + q][c];
  }
  v8h* o = (v8h*)(dst + (size_t)(k0 + c) * NPTS + i0 + rg);
  o[0] = t0;
  o[1] = t1;
}

__global__ void init_out_kernel(float* out) { out[0] = 0.0f; }

// ---------------------------------------------------------------------------
// Main: symmetric 128x128 tiles of sim(i,j)*mask(i,j), upper triangle only
// (off-diagonal tiles weighted x2). 8 waves/block, wave w owns rows
// [bi*128+16w, +16), all 128 cols => 8 v8f accumulators.
// dot is computed as hi.hi + hi.lo + lo.hi in f16 WMMA with f32 accumulate.
// ---------------------------------------------------------------------------
__global__ __launch_bounds__(256) void gram_sim_kernel(
    const f16* __restrict__ xhi, const f16* __restrict__ xlo,
    const f16* __restrict__ xThi, const f16* __restrict__ xTlo,
    const float* __restrict__ sq, const float* __restrict__ mask,
    float* __restrict__ out) {
  const int bi = blockIdx.x;
  const int bj = blockIdx.y;
  if (bj < bi) return;  // block-uniform: EXEC remains all-ones for WMMA

  const int tid  = threadIdx.x;
  const int wv   = tid >> 5;
  const int lane = tid & 31;
  const int ln   = lane & 15;   // column / row-in-tile index
  const int lh   = lane >> 4;   // K-half (A) / M-half (C) selector
  const int iStrip = bi * 128 + wv * 16;
  const int jBase  = bj * 128;

  __shared__ f16 shHi[32 * BPAD];  // B slab: [k][n], k=0..31, n=0..127
  __shared__ f16 shLo[32 * BPAD];
  __shared__ float red[256];

  // Prefetch this block's 128x128 mask tile (64 KB) into cache so the
  // epilogue's HBM-bound mask stream overlaps the 192 WMMAs of GEMM work.
  // Emits global_prefetch_b8; two 256B segments per row, one per thread-pair.
  {
    const int pr = tid >> 1;          // 0..127 mask row within tile
    const int ps = (tid & 1) * 64;    // 0 or 64 floats (256 B)
    const float* p = mask + (size_t)(bi * 128 + pr) * NPTS + jBase + ps;
    __builtin_prefetch(p, 0, 1);
  }

  v8f acc[8];
#pragma unroll
  for (int t = 0; t < 8; ++t)
#pragma unroll
    for (int r = 0; r < 8; ++r) acc[t][r] = 0.0f;

  // A fragment base: lane ln = row M, lh selects K-halves (K=8*lh and 16+8*lh)
  const f16* aRowHi = xhi + (size_t)(iStrip + ln) * DFEAT + 8 * lh;
  const f16* aRowLo = xlo + (size_t)(iStrip + ln) * DFEAT + 8 * lh;

  // Staging map: thread -> (k-row, 16-col chunk) of the 32x128 slab
  const int sr = tid >> 3;
  const int sc = (tid & 7) * 16;

  for (int k0 = 0; k0 < DFEAT; k0 += 32) {
    __syncthreads();
    {
      // xT is [D][N]: contiguous global reads, contiguous LDS writes
      const v8h* gH = (const v8h*)(xThi + (size_t)(k0 + sr) * NPTS + jBase + sc);
      const v8h* gL = (const v8h*)(xTlo + (size_t)(k0 + sr) * NPTS + jBase + sc);
      v8h h0 = gH[0], h1 = gH[1], l0 = gL[0], l1 = gL[1];
      *(v8h*)(shHi + sr * BPAD + sc)     = h0;
      *(v8h*)(shHi + sr * BPAD + sc + 8) = h1;
      *(v8h*)(shLo + sr * BPAD + sc)     = l0;
      *(v8h*)(shLo + sr * BPAD + sc + 8) = l1;
    }
    __syncthreads();

    v16h aHi = frag_from2(aRowHi + k0, aRowHi + k0 + 16);
    v16h aLo = frag_from2(aRowLo + k0, aRowLo + k0 + 16);

#pragma unroll
    for (int t = 0; t < 8; ++t) {
      // B layout: lane ln = N-column, lh selects K-halves (K=8*lh+s and 16+8*lh+s)
      const f16* pH = shHi + t * 16 + ln;
      const f16* pL = shLo + t * 16 + ln;
      v16h bHi, bLo;
#pragma unroll
      for (int s = 0; s < 8; ++s) {
        bHi[s]     = pH[(8 * lh + s) * BPAD];
        bHi[s + 8] = pH[(16 + 8 * lh + s) * BPAD];
        bLo[s]     = pL[(8 * lh + s) * BPAD];
        bLo[s + 8] = pL[(16 + 8 * lh + s) * BPAD];
      }
      acc[t] = __builtin_amdgcn_wmma_f32_16x16x32_f16(false, aHi, false, bHi,
                                                      (short)0, acc[t], false, false);
      acc[t] = __builtin_amdgcn_wmma_f32_16x16x32_f16(false, aHi, false, bLo,
                                                      (short)0, acc[t], false, false);
      acc[t] = __builtin_amdgcn_wmma_f32_16x16x32_f16(false, aLo, false, bHi,
                                                      (short)0, acc[t], false, false);
    }
  }

  // Epilogue. C/D layout: VGPR r, lane -> M = r + 8*lh, N = ln.
  float sqi[8];
#pragma unroll
  for (int r = 0; r < 8; ++r) sqi[r] = sq[iStrip + r + 8 * lh];

  // One base pointer; all 64 mask loads become immediate-offset loads.
  const float* mbase = mask + (size_t)(iStrip + 8 * lh) * NPTS + jBase + ln;

  const float LOG2E = 1.4426950408889634f;
  const float LN2   = 0.6931471805599453f;

  float local = 0.0f;
#pragma unroll
  for (int t = 0; t < 8; ++t) {
    const int j = jBase + t * 16 + ln;
    const float sqj = sq[j];
#pragma unroll
    for (int r = 0; r < 8; ++r) {
      float d2 = sqi[r] + sqj - 2.0f * acc[t][r];
      d2 = fmaxf(d2, 0.0f);
      float dist = __builtin_amdgcn_sqrtf(d2);            // v_sqrt_f32
      float z = 5.0f * (1.0f - dist);
      // softplus(z) = max(z,0) + ln2 * log2(1 + exp2(-|z|*log2e))
      float e  = __builtin_amdgcn_exp2f(-fabsf(z) * LOG2E);  // v_exp_f32
      float lg = __builtin_amdgcn_logf(1.0f + e);            // v_log_f32
      float sp = fmaxf(z, 0.0f) + LN2 * lg;
      float sim = 0.4f * sp - 1.0027f;
      local += mbase[(size_t)r * NPTS + t * 16] * sim;    // coalesced mask stream
    }
  }
  if (bi != bj) local *= 2.0f;  // symmetric off-diagonal tiles counted twice

  red[tid] = local;
  __syncthreads();
#pragma unroll
  for (int o = 128; o > 0; o >>= 1) {
    if (tid < o) red[tid] += red[tid + o];
    __syncthreads();
  }
  if (tid == 0) atomicAdd(out, red[0] * (1.0f / (float)NPTS));
}

// ---------------------------------------------------------------------------
extern "C" void kernel_launch(void* const* d_in, const int* in_sizes, int n_in,
                              void* d_out, int out_size, void* d_ws, size_t ws_size,
                              hipStream_t stream) {
  const float* images = (const float*)d_in[0];       // [9216, 1, 16, 16] f32
  const float* mask   = (const float*)d_in[1];       // [9216, 9216] f32
  float* out = (float*)d_out;                        // scalar f32

  // Workspace layout (~18.9 MB total):
  char* ws = (char*)d_ws;
  const size_t xbytes = (size_t)NPTS * DFEAT * sizeof(f16);  // 4,718,592 B
  f16* xhi  = (f16*)(ws);
  f16* xlo  = (f16*)(ws + xbytes);
  f16* xThi = (f16*)(ws + 2 * xbytes);
  f16* xTlo = (f16*)(ws + 3 * xbytes);
  float* sq = (float*)(ws + 4 * xbytes);

  prep_kernel<<<NPTS, 256, 0, stream>>>(images, xhi, xlo, sq);
  transpose_kernel<<<dim3(NPTS / 64, DFEAT / 64), 256, 0, stream>>>(xhi, xThi);
  transpose_kernel<<<dim3(NPTS / 64, DFEAT / 64), 256, 0, stream>>>(xlo, xTlo);
  init_out_kernel<<<1, 1, 0, stream>>>(out);
  gram_sim_kernel<<<dim3(NPTS / 128, NPTS / 128), 256, 0, stream>>>(
      xhi, xlo, xThi, xTlo, sq, mask, out);

  (void)in_sizes; (void)n_in; (void)out_size; (void)ws_size;
}
